// PlantST_model_37074157699314
// MI455X (gfx1250) — matrix operand order repaired and checked by reading, not compiled, not run on hardware
//
#include <hip/hip_runtime.h>
#include <math.h>

// ---------------------------------------------------------------------------
// CDNA5 (gfx1250) PlantST inference pipeline.
// All dense GEMMs go through v_wmma_f32_16x16x32_bf16 (wave32 WMMA),
// software-pipelined with LDS double buffering; 32 output rows per wave so
// each B fragment ds_load feeds two WMMAs.
// ---------------------------------------------------------------------------

typedef __attribute__((ext_vector_type(16))) __bf16 v16bf;
typedef __attribute__((ext_vector_type(8)))  __bf16 v8bf;
typedef __attribute__((ext_vector_type(8)))  float  v8f;

union AFrag { v16bf v; v8bf h[2]; };

// ---------------------------------------------------------------------------
// Generic tiled GEMM with fused epilogue.
//   C[M,N] = epi( A[M,K](f32) @ W[K,N](f32) + bias )
// EPI: 0 = bias only (bias may be null), 1 = BN + ELU, 2 = BN + ReLU
// Block tile: 256 rows x 64 cols, 256 threads (8 waves).
// Wave w owns rows 32w..32w+31 (two 16-row WMMA strips) and 4 column subtiles.
// Per K-step: 4 A-frag ds_load_b128 + 64 B-element LDS loads  ->  8x v_wmma.
// Pipeline: gload(kt+1) -> compute(kt) -> dstore(kt+1), one barrier/K-step.
// Assumes K % 4 == 0 and N % 4 == 0 (true for every call in this model).
// ---------------------------------------------------------------------------
template <int EPI>
__global__ __launch_bounds__(256) void gemm_bn_act(
    const float* __restrict__ A, const float* __restrict__ W,
    const float* __restrict__ bias,
    const float* __restrict__ bn_g, const float* __restrict__ bn_b,
    const float* __restrict__ bn_m, const float* __restrict__ bn_v,
    float* __restrict__ C, int ldC,
    float* __restrict__ C2, int ldC2,
    int M, int K, int N)
{
    __shared__ __align__(16) __bf16 lA[2][256][40];  // 32 used + pad (80B row = 5x16B)
    __shared__ __align__(16) __bf16 lB[2][32][72];   // 64 used + pad (144B row = 9x16B)

    const int tid  = threadIdx.x;
    const int lane = tid & 31;
    const int wave = tid >> 5;
    const int half = lane >> 4;
    const int mr   = lane & 15;
    const int m0   = blockIdx.x * 256;
    const int n0   = blockIdx.y * 64;

    v8f accA[4] = {};   // rows wave*32 .. +15
    v8f accB[4] = {};   // rows wave*32+16 .. +31

    // A-tile staging: thread t covers the full 32-col slice of row t
    const int arow = tid;
    int agrow = m0 + arow; if (agrow >= M) agrow = M - 1;   // clamp (masked at store)
    const float* aptr = A + (size_t)agrow * (size_t)K;

    // B-tile staging: thread t covers K-row t>>3, cols (t&7)*8 .. +7
    const int brow = tid >> 3;
    const int bcb  = (tid & 7) * 8;
    // loop-invariant N-boundary handling: clamped column + select (no branches)
    const int  nb  = n0 + bcb;
    const bool ok0 = (nb + 3) < N;
    const bool ok1 = (nb + 7) < N;
    const int  ncl = (N >= 4) ? (N - 4) : 0;
    const int  nc0 = ok0 ? nb : ncl;
    const int  nc1 = ok1 ? (nb + 4) : ncl;

    float areg[32];
    float breg[8];

    // ---- issue global loads for K-tile kt into registers ----
    auto gload = [&](int kt) {
        const int k0 = kt << 5;
        if (k0 + 32 <= K) {                       // full tile: unconditional
            #pragma unroll
            for (int j = 0; j < 32; j += 4) {
                const float4 va = *(const float4*)(aptr + k0 + j);
                areg[j+0] = va.x; areg[j+1] = va.y; areg[j+2] = va.z; areg[j+3] = va.w;
            }
            const float* srcr = W + (size_t)(k0 + brow) * (size_t)N;
            const float4 b0v = *(const float4*)(srcr + nc0);
            const float4 b1v = *(const float4*)(srcr + nc1);
            breg[0] = ok0 ? b0v.x : 0.f; breg[1] = ok0 ? b0v.y : 0.f;
            breg[2] = ok0 ? b0v.z : 0.f; breg[3] = ok0 ? b0v.w : 0.f;
            breg[4] = ok1 ? b1v.x : 0.f; breg[5] = ok1 ? b1v.y : 0.f;
            breg[6] = ok1 ? b1v.z : 0.f; breg[7] = ok1 ? b1v.w : 0.f;
        } else {                                  // remainder tile (once per GEMM)
            #pragma unroll
            for (int j = 0; j < 32; j += 4) {
                const int kk = k0 + j;
                float4 va = {0.f, 0.f, 0.f, 0.f};
                if (kk + 3 < K) va = *(const float4*)(aptr + kk);  // K%4==0
                areg[j+0] = va.x; areg[j+1] = va.y; areg[j+2] = va.z; areg[j+3] = va.w;
            }
            const int gk = k0 + brow;
            float4 b0v = {0.f, 0.f, 0.f, 0.f};
            float4 b1v = {0.f, 0.f, 0.f, 0.f};
            if (gk < K) {
                const float* srcr = W + (size_t)gk * (size_t)N;
                b0v = *(const float4*)(srcr + nc0);
                b1v = *(const float4*)(srcr + nc1);
            }
            breg[0] = ok0 ? b0v.x : 0.f; breg[1] = ok0 ? b0v.y : 0.f;
            breg[2] = ok0 ? b0v.z : 0.f; breg[3] = ok0 ? b0v.w : 0.f;
            breg[4] = ok1 ? b1v.x : 0.f; breg[5] = ok1 ? b1v.y : 0.f;
            breg[6] = ok1 ? b1v.z : 0.f; breg[7] = ok1 ? b1v.w : 0.f;
        }
    };

    // ---- convert f32 -> bf16, packed ds_store_b128 into LDS buffer ----
    auto dstore = [&](int buf) {
        __bf16* adst = &lA[buf][arow][0];
        #pragma unroll
        for (int c = 0; c < 4; ++c) {
            v8bf pa;
            #pragma unroll
            for (int j = 0; j < 8; ++j) pa[j] = (__bf16)areg[c * 8 + j];
            *(v8bf*)(adst + c * 8) = pa;
        }
        v8bf pb;
        #pragma unroll
        for (int j = 0; j < 8; ++j) pb[j] = (__bf16)breg[j];
        *(v8bf*)(&lB[buf][brow][bcb]) = pb;
    };

    // ---- A-frag ds_load_b128 + B-frag element loads + 8x WMMA ----
    auto compute = [&](int buf) {
        AFrag af0, af1;
        const __bf16* ar0 = &lA[buf][wave * 32 + mr][0];
        const __bf16* ar1 = &lA[buf][wave * 32 + 16 + mr][0];
        af0.h[0] = *(const v8bf*)(ar0 + half * 8);       // K = 8*half + 0..7
        af0.h[1] = *(const v8bf*)(ar0 + 16 + half * 8);  // K = 16 + 8*half + 0..7
        af1.h[0] = *(const v8bf*)(ar1 + half * 8);
        af1.h[1] = *(const v8bf*)(ar1 + 16 + half * 8);

        const __bf16* br = &lB[buf][0][0];               // column n = s*16 + mr, K in table order
        #pragma unroll
        for (int s = 0; s < 4; ++s) {
            AFrag bf;
            #pragma unroll
            for (int j = 0; j < 8; ++j) {
                bf.v[j]     = br[(half * 8 + j) * 72 + s * 16 + mr];
                bf.v[8 + j] = br[(16 + half * 8 + j) * 72 + s * 16 + mr];
            }
            accA[s] = __builtin_amdgcn_wmma_f32_16x16x32_bf16(false, af0.v, false, bf.v, (short)0, accA[s], false, false);
            accB[s] = __builtin_amdgcn_wmma_f32_16x16x32_bf16(false, af1.v, false, bf.v, (short)0, accB[s], false, false);
        }
    };

    const int nk = (K + 31) >> 5;

    // prologue: stage tile 0
    gload(0);
    dstore(0);
    __syncthreads();

    for (int kt = 0; kt < nk; ++kt) {
        const int buf = kt & 1;
        if (kt + 1 < nk) gload(kt + 1);                 // global loads in flight
        if (kt + 2 < nk) __builtin_prefetch(aptr + ((kt + 2) << 5), 0, 1);
        compute(buf);                                   // overlaps with loads
        if (kt + 1 < nk) dstore(buf ^ 1);               // waits loads, fills other buf
        __syncthreads();
    }

    // ---- epilogue: C layout (ISA 7.12.2): VGPR r -> M = r + 8*half, N = lane&15 ----
    #pragma unroll
    for (int p = 0; p < 2; ++p) {
        const int rbase = m0 + wave * 32 + p * 16 + 8 * half;
        #pragma unroll
        for (int s = 0; s < 4; ++s) {
            const v8f accv = p ? accB[s] : accA[s];
            const int n = n0 + s * 16 + mr;
            if (n >= N) continue;
            const float bval = bias ? bias[n] : 0.0f;
            float scale = 1.0f, shift = bval;
            if (EPI >= 1) {
                const float rs = rsqrtf(bn_v[n] + 1e-3f);
                scale = rs * bn_g[n];
                shift = (bval - bn_m[n]) * scale + bn_b[n];
            }
            #pragma unroll
            for (int r = 0; r < 8; ++r) {
                const int rowi = rbase + r;
                if (rowi >= M) continue;
                float xv = accv[r] * scale + shift;
                if (EPI == 1) xv = (xv > 0.0f) ? xv : (expf(xv) - 1.0f);  // ELU
                if (EPI == 2) xv = fmaxf(xv, 0.0f);                       // ReLU
                C[(size_t)rowi * (size_t)ldC + n] = xv;
                if (C2) C2[(size_t)rowi * (size_t)ldC2 + n] = xv;
            }
        }
    }
}

// ---------------------------------------------------------------------------
// Elementwise / graph kernels
// ---------------------------------------------------------------------------
__global__ void fill_kernel(float* __restrict__ p, float val, long long n) {
    long long i = (long long)blockIdx.x * blockDim.x + threadIdx.x;
    if (i < n) p[i] = val;
}

__global__ void degree_kernel(const int* __restrict__ col, float* __restrict__ deg, int E) {
    int e = blockIdx.x * blockDim.x + threadIdx.x;
    if (e < E) atomicAdd(&deg[col[e]], 1.0f);
}

__global__ void norm_kernel(const float* __restrict__ deg, float* __restrict__ dinv,
                            float* __restrict__ selfw, int n) {
    int i = blockIdx.x * blockDim.x + threadIdx.x;
    if (i < n) {
        const float d = deg[i];
        dinv[i]  = rsqrtf(d);
        selfw[i] = 1.0f / d;
    }
}

// scatter-add of per-edge scaled messages; width floats/row, lgch = log2(width/4)
__global__ __launch_bounds__(256) void scatter_kernel(
    const int* __restrict__ row_idx, const int* __restrict__ col_idx,
    const float* __restrict__ dinv, const float* __restrict__ src,
    float* __restrict__ dst, int E, int width, int lgch)
{
    long long t = (long long)blockIdx.x * blockDim.x + threadIdx.x;
    const int e = (int)(t >> lgch);
    const int f = ((int)t & ((1 << lgch) - 1)) * 4;
    if (e >= E) return;
    const int r = row_idx[e];
    const int c = col_idx[e];
    const float w = dinv[r] * dinv[c];
    const float4 m = *(const float4*)(src + (size_t)r * width + f);
    float* d = dst + (size_t)c * width + f;
    atomicAdd(d + 0, m.x * w);
    atomicAdd(d + 1, m.y * w);
    atomicAdd(d + 2, m.z * w);
    atomicAdd(d + 3, m.w * w);
}

// c = relu(bn(agg + hw*selfw + conv_b))   [in-place over agg buffer], width 128
__global__ void finalize_conv_kernel(
    const float* __restrict__ agg, const float* __restrict__ hw,
    const float* __restrict__ selfw, const float* __restrict__ b,
    const float* __restrict__ g, const float* __restrict__ bb,
    const float* __restrict__ m, const float* __restrict__ v,
    float* __restrict__ c, long long total)
{
    long long i = (long long)blockIdx.x * blockDim.x + threadIdx.x;
    if (i >= total) return;
    const int node = (int)(i >> 7);
    const int f = (int)i & 127;
    float x = agg[i] + hw[i] * selfw[node] + b[f];
    const float rs = rsqrtf(v[f] + 1e-3f);
    x = (x - m[f]) * rs * g[f] + bb[f];
    c[i] = fmaxf(x, 0.0f);
}

// mu/logvar finalize over 32-wide fused buffer (cols 0-15 mu, 16-31 logvar).
// mu is also gnn_z and z[:,64:80].
__global__ void finalize_ml_kernel(
    const float* __restrict__ aggml, const float* __restrict__ mwlv,
    const float* __restrict__ selfw,
    const float* __restrict__ mean_b, const float* __restrict__ logvar_b,
    float* __restrict__ z80, float* __restrict__ mu_out,
    float* __restrict__ lv_out, float* __restrict__ gz_out, long long total)
{
    long long i = (long long)blockIdx.x * blockDim.x + threadIdx.x;
    if (i >= total) return;
    const int node = (int)(i >> 5);
    const int f = (int)i & 31;
    float x = aggml[i] + mwlv[i] * selfw[node];
    if (f < 16) {
        x += mean_b[f];
        mu_out[(size_t)node * 16 + f] = x;
        gz_out[(size_t)node * 16 + f] = x;
        z80[(size_t)node * 80 + 64 + f] = x;
    } else {
        x += logvar_b[f - 16];
        lv_out[(size_t)node * 16 + (f - 16)] = x;
    }
}

// student-t soft assignment: q[n,k], K=15 clusters of dim 80
__global__ __launch_bounds__(256) void cluster_kernel(
    const float* __restrict__ z, const float* __restrict__ cl,
    float* __restrict__ q, int Nn)
{
    __shared__ float scl[15 * 80];
    for (int i = threadIdx.x; i < 15 * 80; i += blockDim.x) scl[i] = cl[i];
    __syncthreads();
    const int node = blockIdx.x * blockDim.x + threadIdx.x;
    if (node >= Nn) return;
    const float* zr = z + (size_t)node * 80;
    float zsq = 0.0f;
    for (int j = 0; j < 80; ++j) { const float t = zr[j]; zsq += t * t; }
    float qv[15];
    float sum = 0.0f;
    #pragma unroll
    for (int k = 0; k < 15; ++k) {
        const float* cr = scl + k * 80;
        float dot = 0.0f, cn = 0.0f;
        for (int j = 0; j < 80; ++j) { const float c = cr[j]; dot += zr[j] * c; cn += c * c; }
        const float dist = fmaxf(zsq + cn - 2.0f * dot, 0.0f);
        const float t = 1.0f / (1.0f + dist / 0.9f + 1e-8f);
        const float qq = powf(t, 0.95f);          // (ALPHA+1)/2 = 0.95
        qv[k] = qq; sum += qq;
    }
    const float inv = 1.0f / sum;
    #pragma unroll
    for (int k = 0; k < 15; ++k) q[(size_t)node * 15 + k] = qv[k] * inv;
}

// ---------------------------------------------------------------------------
// Host side
// ---------------------------------------------------------------------------
static void launch_gemm(int epi,
                        const float* A, const float* W, const float* bias,
                        const float* g, const float* bb, const float* m, const float* v,
                        float* C, int ldC, float* C2, int ldC2,
                        int M, int K, int N, hipStream_t s)
{
    dim3 grid((M + 255) / 256, (N + 63) / 64);
    if (epi == 0)
        gemm_bn_act<0><<<grid, 256, 0, s>>>(A, W, bias, g, bb, m, v, C, ldC, C2, ldC2, M, K, N);
    else if (epi == 1)
        gemm_bn_act<1><<<grid, 256, 0, s>>>(A, W, bias, g, bb, m, v, C, ldC, C2, ldC2, M, K, N);
    else
        gemm_bn_act<2><<<grid, 256, 0, s>>>(A, W, bias, g, bb, m, v, C, ldC, C2, ldC2, M, K, N);
}

extern "C" void kernel_launch(void* const* d_in, const int* in_sizes, int n_in,
                              void* d_out, int out_size, void* d_ws, size_t ws_size,
                              hipStream_t stream)
{
    const int DIN = 1000;
    const int Nn  = in_sizes[0] / DIN;       // 100000 nodes
    const int E   = in_sizes[1] / 2;         // 1600000 edges

    const float* x      = (const float*)d_in[0];
    const int*   ei     = (const int*)d_in[1];
    const int*   row    = ei;
    const int*   col    = ei + E;
    const float* enc0_w = (const float*)d_in[2];
    const float* enc0_b = (const float*)d_in[3];
    const float* enc0_g = (const float*)d_in[4];
    const float* enc0_bb= (const float*)d_in[5];
    const float* enc0_m = (const float*)d_in[6];
    const float* enc0_v = (const float*)d_in[7];
    const float* enc1_w = (const float*)d_in[8];
    const float* enc1_b = (const float*)d_in[9];
    const float* enc1_g = (const float*)d_in[10];
    const float* enc1_bb= (const float*)d_in[11];
    const float* enc1_m = (const float*)d_in[12];
    const float* enc1_v = (const float*)d_in[13];
    const float* conv_w = (const float*)d_in[14];
    const float* conv_b = (const float*)d_in[15];
    const float* conv_g = (const float*)d_in[16];
    const float* conv_bb= (const float*)d_in[17];
    const float* conv_m = (const float*)d_in[18];
    const float* conv_v = (const float*)d_in[19];
    const float* mean_w = (const float*)d_in[20];
    const float* mean_b = (const float*)d_in[21];
    const float* lv_w   = (const float*)d_in[22];
    const float* lv_b   = (const float*)d_in[23];
    const float* dec0_w = (const float*)d_in[24];
    const float* dec0_b = (const float*)d_in[25];
    const float* dec0_g = (const float*)d_in[26];
    const float* dec0_bb= (const float*)d_in[27];
    const float* dec0_m = (const float*)d_in[28];
    const float* dec0_v = (const float*)d_in[29];
    const float* dec1_w = (const float*)d_in[30];
    const float* dec1_b = (const float*)d_in[31];
    const float* clus   = (const float*)d_in[32];

    // output layout (flat, return order): z, mu, logvar, de_feat, q, feat_x, gnn_z
    float* out = (float*)d_out;
    float* oZ  = out;
    float* oMU = oZ  + (size_t)Nn * 80;
    float* oLV = oMU + (size_t)Nn * 16;
    float* oDE = oLV + (size_t)Nn * 16;
    float* oQ  = oDE + (size_t)Nn * 1000;
    float* oFX = oQ  + (size_t)Nn * 15;
    float* oGZ = oFX + (size_t)Nn * 64;

    // workspace layout (floats); dbuf reuses h's region (h dead by then)
    float* ws = (float*)d_ws;
    float* w_h     = ws;                            // Nn*256
    float* w_d     = ws;                            // Nn*64  (reuse of h)
    float* w_hw    = w_h     + (size_t)Nn * 256;    // Nn*128
    float* w_agg   = w_hw    + (size_t)Nn * 128;    // Nn*128 (-> c in place)
    float* w_mwlv  = w_agg   + (size_t)Nn * 128;    // Nn*32
    float* w_aggml = w_mwlv  + (size_t)Nn * 32;     // Nn*32
    float* w_deg   = w_aggml + (size_t)Nn * 32;     // Nn
    float* w_dinv  = w_deg   + (size_t)Nn;          // Nn
    float* w_self  = w_dinv  + (size_t)Nn;          // Nn

    const int TB = 256;
    #define GRID1(n) dim3((unsigned)(((long long)(n) + TB - 1) / TB))

    // 1) h = elu(bn(x @ enc0_w + b))                       [Nn,256]
    launch_gemm(1, x, enc0_w, enc0_b, enc0_g, enc0_bb, enc0_m, enc0_v,
                w_h, 256, nullptr, 0, Nn, DIN, 256, stream);

    // 2) feat_x = elu(bn(h @ enc1_w + b)) -> oFX and z[:,0:64]
    launch_gemm(1, w_h, enc1_w, enc1_b, enc1_g, enc1_bb, enc1_m, enc1_v,
                oFX, 64, oZ, 80, Nn, 256, 64, stream);

    // 3) graph normalization: deg = 1 + indegree; dinv = deg^-1/2; selfw = 1/deg
    fill_kernel<<<GRID1(Nn), TB, 0, stream>>>(w_deg, 1.0f, Nn);
    fill_kernel<<<GRID1((long long)Nn * 128), TB, 0, stream>>>(w_agg, 0.0f, (long long)Nn * 128);
    fill_kernel<<<GRID1((long long)Nn * 32), TB, 0, stream>>>(w_aggml, 0.0f, (long long)Nn * 32);
    degree_kernel<<<GRID1(E), TB, 0, stream>>>(col, w_deg, E);
    norm_kernel<<<GRID1(Nn), TB, 0, stream>>>(w_deg, w_dinv, w_self, Nn);

    // 4) hw = feat_x @ conv_w                              [Nn,128]
    launch_gemm(0, oFX, conv_w, nullptr, nullptr, nullptr, nullptr, nullptr,
                w_hw, 128, nullptr, 0, Nn, 64, 128, stream);

    // 5) agg[col] += hw[row] * dinv[row]*dinv[col]   (128 feats, float4 chunks)
    scatter_kernel<<<GRID1((long long)E * 32), TB, 0, stream>>>(row, col, w_dinv,
                                                                w_hw, w_agg, E, 128, 5);

    // 6) c = relu(bn(agg + hw/deg + conv_b))  (in-place into agg buffer)
    finalize_conv_kernel<<<GRID1((long long)Nn * 128), TB, 0, stream>>>(
        w_agg, w_hw, w_self, conv_b, conv_g, conv_bb, conv_m, conv_v,
        w_agg, (long long)Nn * 128);

    // 7) project BEFORE scatter (edge weight is scalar): mwlv = c @ [mean_w | lv_w]
    launch_gemm(0, w_agg, mean_w, nullptr, nullptr, nullptr, nullptr, nullptr,
                w_mwlv, 32, nullptr, 0, Nn, 128, 16, stream);
    launch_gemm(0, w_agg, lv_w, nullptr, nullptr, nullptr, nullptr, nullptr,
                w_mwlv + 16, 32, nullptr, 0, Nn, 128, 16, stream);

    // 8) scatter both heads (32 feats/edge)
    scatter_kernel<<<GRID1((long long)E * 8), TB, 0, stream>>>(row, col, w_dinv,
                                                               w_mwlv, w_aggml, E, 32, 3);

    // 9) finalize mu/logvar; mu -> z[:,64:80], oMU, oGZ; logvar -> oLV
    finalize_ml_kernel<<<GRID1((long long)Nn * 32), TB, 0, stream>>>(
        w_aggml, w_mwlv, w_self, mean_b, lv_b, oZ, oMU, oLV, oGZ, (long long)Nn * 32);

    // 10) d = elu(bn(z @ dec0_w + b))                      [Nn,64]  (K=80)
    launch_gemm(1, oZ, dec0_w, dec0_b, dec0_g, dec0_bb, dec0_m, dec0_v,
                w_d, 64, nullptr, 0, Nn, 80, 64, stream);

    // 11) de_feat = d @ dec1_w + dec1_b                    [Nn,1000] -> oDE
    launch_gemm(0, w_d, dec1_w, dec1_b, nullptr, nullptr, nullptr, nullptr,
                oDE, 1000, nullptr, 0, Nn, 64, 1000, stream);

    // 12) student-t soft assignment q                      [Nn,15] -> oQ
    cluster_kernel<<<GRID1(Nn), TB, 0, stream>>>(oZ, clus, oQ, Nn);

    #undef GRID1
    (void)n_in; (void)out_size; (void)ws_size;
}
